// SelfAttentionBlock_15573551415768
// MI455X (gfx1250) — matrix operand order repaired and checked by reading, not compiled, not run on hardware
//
#include <hip/hip_runtime.h>
#include <hip/hip_bf16.h>

// ---------------- problem constants ----------------
#define BATCH 64
#define CFULL 4096        // C
#define NDIM  256         // N == CQ
#define KT    32          // K slab staged through LDS per iteration
#define SA    40          // LDS stride (bf16 elems) for staging tiles: 80 bytes, 16B-aligned rows
#define SV    264         // LDS stride for V tile: 528 bytes = 16*33, 16B-aligned rows

typedef __attribute__((ext_vector_type(16))) __bf16 v16bf;
typedef __attribute__((ext_vector_type(8)))  float  v8f;

union Frag {
    v16bf v;
    int4  q[2];
    unsigned int u[8];
};

__device__ inline unsigned short f2bf(float f) {
    unsigned int u = __float_as_uint(f);
    u += 0x7fffu + ((u >> 16) & 1u);        // round-to-nearest-even
    return (unsigned short)(u >> 16);
}

// A fragment (16x32 bf16, MxK) from a row-major [row][k] array, stride in elems.
// ISA layout: lanes 0-15 = rows, VGPR0-3 = K[koff..koff+7], VGPR4-7 = K[koff+16..koff+23],
// koff = (lane>>4)*8.
__device__ inline Frag load_frag_a(const unsigned short* base, int stride, int row0, int k0) {
    int lane = threadIdx.x & 31;
    int r  = row0 + (lane & 15);
    int kh = k0 + ((lane >> 4) << 3);
    const int4* p = (const int4*)(base + (size_t)r * stride + kh);
    Frag f;
    f.q[0] = p[0];      // K = kh .. kh+7
    f.q[1] = p[2];      // K = kh+16 .. kh+23  (+32 bytes)
    return f;
}

// B fragment (32x16 bf16, KxN) from a TRANSPOSED row-major [n][k] array.
// ISA layout: lanes 0-15 = cols, VGPR0-3 = K[koff..koff+7], VGPR4-7 = K[koff+16..koff+23], koff = (lane>>4)*8.
__device__ inline Frag load_frag_b(const unsigned short* base, int stride, int n0, int k0) {
    int lane = threadIdx.x & 31;
    int n  = n0 + (lane & 15);
    int kb = k0 + ((lane >> 4) << 3);
    const int4* p = (const int4*)(base + (size_t)n * stride + kb);
    Frag f;
    f.q[0] = p[0];      // K = kb .. kb+7
    f.q[1] = p[2];      // K = kb+16 .. kb+23  (+32 bytes)
    return f;
}

__device__ inline v8f wmma_bf16(const Frag& a, const Frag& b, v8f c) {
    return __builtin_amdgcn_wmma_f32_16x16x32_bf16(
        /*neg_a=*/false, a.v, /*neg_b=*/false, b.v,
        /*c_mod=*/(short)0, c, /*reuse_a=*/false, /*reuse_b=*/false);
}

__device__ inline v8f vzero() {
    v8f z;
    #pragma unroll
    for (int i = 0; i < 8; ++i) z[i] = 0.0f;
    return z;
}

// ---------------------------------------------------------------------------
// Kernel 1: Out_bf16[b, m, n] = sum_c W[m,c] * X[b,c,n] + bias[m]
// Block = 256 threads (8 waves), tile = 64 rows x 256 cols, K staged 32-deep.
// grid = (Mrows/64, BATCH)
// ---------------------------------------------------------------------------
__global__ __launch_bounds__(256) void k_proj64(
    const float* __restrict__ W,      // [Mrows, 4096] row-major
    const float* __restrict__ bias,   // [Mrows]
    const float* __restrict__ X,      // [B, 4096, 256]
    unsigned short* __restrict__ Out, // bf16 [B, Mrows, 256]
    int Mrows)
{
    __shared__ unsigned short ldsA[64 * SA];   // W slab  [r][k]
    __shared__ unsigned short ldsB[NDIM * SA]; // X slab transposed [n][k]

    const int b   = blockIdx.y;
    const int m0  = blockIdx.x * 64;
    const int tid = threadIdx.x;
    const int wave = tid >> 5;
    const int r0 = (wave & 3) << 4;   // 0,16,32,48
    const int c0 = (wave >> 2) << 7;  // 0 or 128
    const float* Xb = X + (size_t)b * CFULL * NDIM;

    v8f acc[8];
    #pragma unroll
    for (int t = 0; t < 8; ++t) acc[t] = vzero();

    for (int kt = 0; kt < CFULL; kt += KT) {
        // stage A: 64 x 32, coalesced over k
        #pragma unroll
        for (int i = 0; i < 8; ++i) {
            int idx = tid + (i << 8);
            int r = idx >> 5, k = idx & 31;
            ldsA[r * SA + k] = f2bf(W[(size_t)(m0 + r) * CFULL + kt + k]);
        }
        // stage B transposed: X[kt+i][n] -> ldsB[n][i], coalesced over n
        #pragma unroll 8
        for (int i = 0; i < KT; ++i) {
            ldsB[tid * SA + i] = f2bf(Xb[(size_t)(kt + i) * NDIM + tid]);
        }
        __syncthreads();

        Frag a = load_frag_a(ldsA, SA, r0, 0);
        #pragma unroll
        for (int t = 0; t < 8; ++t) {
            Frag bb = load_frag_b(ldsB, SA, c0 + (t << 4), 0);
            acc[t] = wmma_bf16(a, bb, acc[t]);
        }
        __syncthreads();
    }

    // epilogue: bias add, convert bf16, store
    const int lane  = tid & 31;
    const int col   = lane & 15;
    const int rbase = (lane >> 4) << 3;
    #pragma unroll
    for (int t = 0; t < 8; ++t) {
        int cc = c0 + (t << 4) + col;
        #pragma unroll
        for (int v = 0; v < 8; ++v) {
            int rr = m0 + r0 + rbase + v;
            float val = acc[t][v] + bias[rr];
            Out[((size_t)b * Mrows + rr) * NDIM + cc] = f2bf(val);
        }
    }
}

// ---------------------------------------------------------------------------
// Kernel 2a: E[b,i,j] = sum_n Q[b,i,n] * K[b,j,n]   (fp32 out)
// Q,K bf16 row-major [256][256]; K's row-major layout IS the [col][k] layout
// the B fragment wants, so no staging needed (both live in L2).
// grid = (4, BATCH)
// ---------------------------------------------------------------------------
__global__ __launch_bounds__(256) void k_energy(
    const unsigned short* __restrict__ Q,
    const unsigned short* __restrict__ Km,
    float* __restrict__ E)
{
    const int b   = blockIdx.y;
    const int m0  = blockIdx.x * 64;
    const int tid = threadIdx.x;
    const int wave = tid >> 5;
    const int r0 = m0 + ((wave & 3) << 4);
    const int c0 = (wave >> 2) << 7;
    const unsigned short* Qb = Q  + (size_t)b * NDIM * NDIM;
    const unsigned short* Kb = Km + (size_t)b * NDIM * NDIM;

    v8f acc[8];
    #pragma unroll
    for (int t = 0; t < 8; ++t) acc[t] = vzero();

    for (int kt = 0; kt < NDIM; kt += 32) {
        Frag a = load_frag_a(Qb, NDIM, r0, kt);
        #pragma unroll
        for (int t = 0; t < 8; ++t) {
            Frag bb = load_frag_b(Kb, NDIM, c0 + (t << 4), kt);
            acc[t] = wmma_bf16(a, bb, acc[t]);
        }
    }

    const int lane  = tid & 31;
    const int col   = lane & 15;
    const int rbase = (lane >> 4) << 3;
    float* Eb = E + (size_t)b * NDIM * NDIM;
    #pragma unroll
    for (int t = 0; t < 8; ++t) {
        int cc = c0 + (t << 4) + col;
        #pragma unroll
        for (int v = 0; v < 8; ++v)
            Eb[(size_t)(r0 + rbase + v) * NDIM + cc] = acc[t][v];
    }
}

// ---------------------------------------------------------------------------
// Kernel 2b: row softmax over 256 elems; one wave32 per row; bf16 out.
// grid = B*256/8 blocks of 256 threads.
// ---------------------------------------------------------------------------
__global__ __launch_bounds__(256) void k_softmax(
    const float* __restrict__ E, unsigned short* __restrict__ A)
{
    const int row  = blockIdx.x * 8 + (threadIdx.x >> 5);
    const int lane = threadIdx.x & 31;
    const float* e = E + (size_t)row * NDIM;

    float vals[8];
    float mx = -3.0e38f;
    #pragma unroll
    for (int j = 0; j < 8; ++j) { vals[j] = e[lane + (j << 5)]; mx = fmaxf(mx, vals[j]); }
    #pragma unroll
    for (int s = 16; s > 0; s >>= 1) mx = fmaxf(mx, __shfl_xor(mx, s, 32));

    float sum = 0.0f;
    #pragma unroll
    for (int j = 0; j < 8; ++j) { vals[j] = __expf(vals[j] - mx); sum += vals[j]; }
    #pragma unroll
    for (int s = 16; s > 0; s >>= 1) sum += __shfl_xor(sum, s, 32);

    float inv = 1.0f / sum;
    #pragma unroll
    for (int j = 0; j < 8; ++j)
        A[(size_t)row * NDIM + lane + (j << 5)] = f2bf(vals[j] * inv);
}

// ---------------------------------------------------------------------------
// Kernel 3 (fused): Vtile = Wv[m0:m0+64,:] * X_b + bv  (WMMA, into LDS bf16),
// then Out[b, m0+r, i] = sum_n Vtile[r,n] * attn[b,i,n]  (WMMA from LDS + global).
// The 268MB `v` tensor never touches HBM. grid = (64, BATCH)
// ---------------------------------------------------------------------------
__global__ __launch_bounds__(256) void k_out(
    const float* __restrict__ Wv,             // [4096,4096]
    const float* __restrict__ bv,             // [4096]
    const float* __restrict__ X,              // [B,4096,256]
    const unsigned short* __restrict__ Attn,  // bf16 [B,256,256]
    float* __restrict__ Out)                  // [B,4096,256]
{
    __shared__ unsigned short ldsA[64 * SA];
    __shared__ unsigned short ldsB[NDIM * SA];
    __shared__ unsigned short ldsV[64 * SV];  // V tile [r][n], bf16

    const int b   = blockIdx.y;
    const int m0  = blockIdx.x * 64;
    const int tid = threadIdx.x;
    const int wave = tid >> 5;
    const int r0 = (wave & 3) << 4;
    const int c0 = (wave >> 2) << 7;
    const float* Xb = X + (size_t)b * CFULL * NDIM;

    // ---- phase 1: V tile = Wv * X + bv ----
    v8f acc[8];
    #pragma unroll
    for (int t = 0; t < 8; ++t) acc[t] = vzero();

    for (int kt = 0; kt < CFULL; kt += KT) {
        #pragma unroll
        for (int i = 0; i < 8; ++i) {
            int idx = tid + (i << 8);
            int r = idx >> 5, k = idx & 31;
            ldsA[r * SA + k] = f2bf(Wv[(size_t)(m0 + r) * CFULL + kt + k]);
        }
        #pragma unroll 8
        for (int i = 0; i < KT; ++i) {
            ldsB[tid * SA + i] = f2bf(Xb[(size_t)(kt + i) * NDIM + tid]);
        }
        __syncthreads();

        Frag a = load_frag_a(ldsA, SA, r0, 0);
        #pragma unroll
        for (int t = 0; t < 8; ++t) {
            Frag bb = load_frag_b(ldsB, SA, c0 + (t << 4), 0);
            acc[t] = wmma_bf16(a, bb, acc[t]);
        }
        __syncthreads();
    }

    // write V tile (+bias) to LDS as bf16
    const int lane  = tid & 31;
    const int col   = lane & 15;
    const int rbase = (lane >> 4) << 3;
    #pragma unroll
    for (int t = 0; t < 8; ++t) {
        int cc = c0 + (t << 4) + col;
        #pragma unroll
        for (int v = 0; v < 8; ++v) {
            int rr = r0 + rbase + v;
            ldsV[rr * SV + cc] = f2bf(acc[t][v] + bv[m0 + rr]);
        }
    }
    __syncthreads();

    // ---- phase 2: Out tile = Vtile * attn^T ----
    // out[c,i] = sum_n V[c,n] attn[i,n]; attn row-major [i][n] is exactly the
    // [col][k] layout the B fragment wants.
    v8f oacc[8];
    #pragma unroll
    for (int t = 0; t < 8; ++t) oacc[t] = vzero();

    const unsigned short* Ab = Attn + (size_t)b * NDIM * NDIM;
    for (int kt = 0; kt < NDIM; kt += 32) {
        Frag a = load_frag_a(ldsV, SV, r0, kt);
        #pragma unroll
        for (int t = 0; t < 8; ++t) {
            Frag bb = load_frag_b(Ab, NDIM, c0 + (t << 4), kt);
            oacc[t] = wmma_bf16(a, bb, oacc[t]);
        }
    }

    float* Ob = Out + (size_t)b * CFULL * NDIM;
    #pragma unroll
    for (int t = 0; t < 8; ++t) {
        int cc = c0 + (t << 4) + col;
        #pragma unroll
        for (int v = 0; v < 8; ++v)
            Ob[(size_t)(m0 + r0 + rbase + v) * NDIM + cc] = oacc[t][v];
    }
}

// ---------------------------------------------------------------------------
extern "C" void kernel_launch(void* const* d_in, const int* in_sizes, int n_in,
                              void* d_out, int out_size, void* d_ws, size_t ws_size,
                              hipStream_t stream) {
    (void)in_sizes; (void)n_in; (void)out_size; (void)ws_size;

    const float* x  = (const float*)d_in[0];  // [64,4096,4,8,8] == [64,4096,256]
    const float* Wq = (const float*)d_in[1];  // [256,4096]
    const float* bq = (const float*)d_in[2];  // [256]
    const float* Wk = (const float*)d_in[3];  // [256,4096]
    const float* bk = (const float*)d_in[4];  // [256]
    const float* Wv = (const float*)d_in[5];  // [4096,4096]
    const float* bv = (const float*)d_in[6];  // [4096]
    float* out = (float*)d_out;               // [64,4096,256]

    // workspace layout (bytes)
    char* ws = (char*)d_ws;
    const size_t szQK = (size_t)BATCH * NDIM * NDIM * sizeof(unsigned short); // 8 MB
    const size_t szE  = (size_t)BATCH * NDIM * NDIM * sizeof(float);          // 16 MB
    unsigned short* Qbuf = (unsigned short*)(ws);
    unsigned short* Kbuf = (unsigned short*)(ws + szQK);
    float*          Ebuf = (float*)(ws + 2 * szQK);
    unsigned short* Abuf = (unsigned short*)(ws + 2 * szQK + szE);

    dim3 blk(256);
    k_proj64<<<dim3(NDIM / 64, BATCH), blk, 0, stream>>>(Wq, bq, x, Qbuf, NDIM);
    k_proj64<<<dim3(NDIM / 64, BATCH), blk, 0, stream>>>(Wk, bk, x, Kbuf, NDIM);
    k_energy<<<dim3(NDIM / 64, BATCH), blk, 0, stream>>>(Qbuf, Kbuf, Ebuf);
    k_softmax<<<dim3(BATCH * NDIM / 8), blk, 0, stream>>>(Ebuf, Abuf);
    k_out<<<dim3(CFULL / 64, BATCH), blk, 0, stream>>>(Wv, bv, x, Abuf, out);
}
